// SourceSpecificGeometricConstraint_62088047231526
// MI455X (gfx1250) — matrix operand order repaired and checked by reading, not compiled, not run on hardware
//
#include <hip/hip_runtime.h>
#include <hip/hip_bf16.h>
#include <math.h>

typedef __attribute__((ext_vector_type(16))) _Float16 v16h;
typedef __attribute__((ext_vector_type(2)))  _Float16 h2v;
typedef __attribute__((ext_vector_type(8)))  float    v8f;

#define S_LEN   160000
#define B_N     16
#define NW      1248                       // number of windows: arange(0, 160000-256, 128)
#define T_INV   (1.0f / 159999.0f)         // linspace(0,1,S) step
#define EN_INV  (1.0f / (1248.0f * 256.0f))// 1 / (nw * WIN)

// ---------------------------------------------------------------------------
// Phase 0: zero the mf accumulator (d_ws)
// ---------------------------------------------------------------------------
__global__ void zero_acc_kernel(float* acc) {
    if (threadIdx.x < B_N) acc[threadIdx.x] = 0.0f;
}

// ---------------------------------------------------------------------------
// Phase 1: mf[b] = weighted sum of mixture^2.
// Overlapping windows (win 256, hop 128) reduce to per-128-block weights:
// block 0 -> 1, blocks 1..1247 -> 2, block 1248 -> 1, blocks >= 1249 -> 0.
// ---------------------------------------------------------------------------
__global__ void mf_reduce_kernel(const float* __restrict__ mix,
                                 float* __restrict__ acc) {
    __shared__ float red[256];
    const int b = blockIdx.y;
    const float* row = mix + (size_t)b * S_LEN;
    float p = 0.0f;
    for (int s = blockIdx.x * blockDim.x + threadIdx.x; s < S_LEN;
         s += gridDim.x * blockDim.x) {
        int j = s >> 7;
        float w = (j > NW) ? 0.0f : ((j == 0 || j == NW) ? 1.0f : 2.0f);
        float x = row[s];
        p = fmaf(w * x, x, p);
    }
    red[threadIdx.x] = p;
    __syncthreads();
    for (int off = 128; off > 0; off >>= 1) {
        if (threadIdx.x < (unsigned)off) red[threadIdx.x] += red[threadIdx.x + off];
        __syncthreads();
    }
    if (threadIdx.x == 0) atomicAdd(&acc[b], red[0]);
}

// ---------------------------------------------------------------------------
// Phase 2: fused gate-MLP + inner-MLP (WMMA) + combine.
// One wave handles 16 consecutive s values:
//   * gate tile:  M = 16 s-values    (A built from t_s * bw1 + bb1)
//   * 16 inner tiles: M = 16 batches (A = relu(t_s*u + (mf_b*v + ib1)))
// Layer 2 (64 -> 32) runs on v_wmma_f32_16x16x32_f16 (K split x2, N split x2).
// Layer 3 (32 -> 1): per-lane dot + 4-round reduce-scatter over 16-lane halves.
// ---------------------------------------------------------------------------
__device__ __forceinline__ v8f wmma_f16(v16h a, v16h b, v8f c) {
    return __builtin_amdgcn_wmma_f32_16x16x32_f16(
        /*neg_a=*/false, a, /*neg_b=*/false, b,
        /*c_mod=*/(short)0, c, /*reuse_a=*/false, /*reuse_b=*/false);
}

// Reduce 16 lanes x 8 values -> each lane ends owning the full sum for
// r = 4*bit3 + 2*bit2 + bit1 of its lane id (duplicated across bit0 pairs).
__device__ __forceinline__ float reduce_scatter16(const float a8[8], int l, int* r_out) {
    const int b3 = (l >> 3) & 1, b2 = (l >> 2) & 1, b1 = (l >> 1) & 1;
    float k4[4];
#pragma unroll
    for (int j = 0; j < 4; ++j) {
        float send = b3 ? a8[j]     : a8[j + 4];
        float keep = b3 ? a8[j + 4] : a8[j];
        k4[j] = keep + __shfl_xor(send, 8, 32);
    }
    float k2[2];
#pragma unroll
    for (int j = 0; j < 2; ++j) {
        float send = b2 ? k4[j]     : k4[j + 2];
        float keep = b2 ? k4[j + 2] : k4[j];
        k2[j] = keep + __shfl_xor(send, 4, 32);
    }
    float send = b1 ? k2[0] : k2[1];
    float keep = b1 ? k2[1] : k2[0];
    float v = keep + __shfl_xor(send, 2, 32);
    v += __shfl_xor(v, 1, 32);
    *r_out = 4 * b3 + 2 * b2 + b1;
    return v;
}

union AFrag { v16h v; h2v p[8]; };

__launch_bounds__(256)
__global__ void fused_mlp_kernel(const float* __restrict__ delta,
                                 const float* __restrict__ acc,
                                 const float* __restrict__ bw1, const float* __restrict__ bb1,
                                 const float* __restrict__ bw2, const float* __restrict__ bb2,
                                 const float* __restrict__ bw3, const float* __restrict__ bb3,
                                 const float* __restrict__ iw1, const float* __restrict__ ib1,
                                 const float* __restrict__ iw2, const float* __restrict__ ib2,
                                 const float* __restrict__ iw3, const float* __restrict__ ib3,
                                 float* __restrict__ out) {
    // LDS: small weights + per-wave staging
    __shared__ float wL[464];
    // wL layout: [0..63]=iw1 row0 (u), [64..127]=iw1 row1 (v), [128..191]=ib1,
    //            [192..255]=bw1, [256..319]=bb1, [320..351]=iw3, [352..383]=bw3,
    //            [384..415]=ib2, [416..447]=bb2, [448..463]=mf
    __shared__ float stage_s0[8][16][16];   // [wave][batch][s-local]
    __shared__ float stage_g[8][16];        // [wave][s-local]

    const int tid = threadIdx.x;
    if (tid < 128) wL[tid] = iw1[tid];
    if (tid < 64) { wL[128 + tid] = ib1[tid]; wL[192 + tid] = bw1[tid]; wL[256 + tid] = bb1[tid]; }
    if (tid < 32) { wL[320 + tid] = iw3[tid]; wL[352 + tid] = bw3[tid];
                    wL[384 + tid] = ib2[tid]; wL[416 + tid] = bb2[tid]; }
    if (tid < 16) { wL[448 + tid] = acc[tid] * EN_INV; }
    __syncthreads();

    const int wv   = tid >> 5;
    const int lane = tid & 31;
    const int n    = lane & 15;   // N column / batch row / s row in tiles
    const int h    = lane >> 4;   // lane half
    const int sbase = (blockIdx.x * 8 + wv) * 16;   // 1250 blocks * 8 waves * 16 = 160000

    // ---- B fragments (f16, K-striped layout: element i -> K = 32*kh + 8*h + i + (i&8))
    v16h iB[2][2], bB[2][2];
#pragma unroll
    for (int kh = 0; kh < 2; ++kh)
#pragma unroll
        for (int nh = 0; nh < 2; ++nh) {
            v16h bi, bb;
#pragma unroll
            for (int i = 0; i < 16; ++i) {
                int k = 32 * kh + 8 * h + i + (i & 8);
                bi[i] = (_Float16)iw2[k * 32 + 16 * nh + n];
                bb[i] = (_Float16)bw2[k * 32 + 16 * nh + n];
            }
            iB[kh][nh] = bi; bB[kh][nh] = bb;
        }

    // ---- per-lane hoisted constants
    const float ib2lo = wL[384 + n],      ib2hi = wL[384 + 16 + n];
    const float bb2lo = wL[416 + n],      bb2hi = wL[416 + 16 + n];
    const float iw3lo = wL[320 + n],      iw3hi = wL[320 + 16 + n];
    const float bw3lo = wL[352 + n],      bw3hi = wL[352 + 16 + n];
    const float ib3s = ib3[0], bb3s = bb3[0];
    const float mfb = wL[448 + n];        // mf for this lane's batch

    // ---- packed layer-1 coefficients for the inner MLP.
    // A-fragment map (16-bit A 16x32): element i -> K = 8*h + i + (i&8),
    // so packed pair j holds (k, k+1) with k = 8*h + 2*j + (j>=4 ? 8 : 0).
    h2v ukp0[8], ukp1[8], ckp0[8], ckp1[8];
#pragma unroll
    for (int j = 0; j < 8; ++j) {
        int k = 8 * h + 2 * j + ((j >= 4) ? 8 : 0);
        ukp0[j] = h2v{ (_Float16)wL[k],      (_Float16)wL[k + 1] };
        ukp1[j] = h2v{ (_Float16)wL[32 + k], (_Float16)wL[32 + k + 1] };
        ckp0[j] = h2v{ (_Float16)fmaf(mfb, wL[64 + k],     wL[128 + k]),
                       (_Float16)fmaf(mfb, wL[64 + k + 1], wL[128 + k + 1]) };
        ckp1[j] = h2v{ (_Float16)fmaf(mfb, wL[96 + k],     wL[160 + k]),
                       (_Float16)fmaf(mfb, wL[96 + k + 1], wL[160 + k + 1]) };
    }
    const h2v zero2 = h2v{ (_Float16)0.0f, (_Float16)0.0f };

    // ================= gate tile: M = 16 s-values =================
    {
        const float tm = (float)(sbase + n) * T_INV;   // t for row m = n
        v16h ga0, ga1;
#pragma unroll
        for (int i = 0; i < 16; ++i) {
            int k = 8 * h + i + (i & 8);
            ga0[i] = (_Float16)fmaxf(fmaf(tm, wL[192 + k],      wL[256 + k]),      0.0f);
            ga1[i] = (_Float16)fmaxf(fmaf(tm, wL[192 + 32 + k], wL[256 + 32 + k]), 0.0f);
        }
        v8f gc0 = {}, gc1 = {};
        gc0 = wmma_f16(ga0, bB[0][0], gc0);
        gc0 = wmma_f16(ga1, bB[1][0], gc0);
        gc1 = wmma_f16(ga0, bB[0][1], gc1);
        gc1 = wmma_f16(ga1, bB[1][1], gc1);

        float a8[8];
#pragma unroll
        for (int r = 0; r < 8; ++r) {
            float x0 = fmaxf(gc0[r] + bb2lo, 0.0f);
            float x1 = fmaxf(gc1[r] + bb2hi, 0.0f);
            a8[r] = fmaf(x0, bw3lo, x1 * bw3hi);
        }
        int rown;
        float v = reduce_scatter16(a8, n, &rown);
        if ((lane & 1) == 0) {
            float z = v + bb3s;
            stage_g[wv][rown + 8 * h] = 1.0f / (1.0f + __expf(-z));
        }
    }

    // ================= 16 inner tiles: M = 16 batches =================
#pragma unroll 1
    for (int sv = 0; sv < 16; ++sv) {
        const float ts = (float)(sbase + sv) * T_INV;
        const _Float16 tsh = (_Float16)ts;
        const h2v ts2 = h2v{ tsh, tsh };

        AFrag a0, a1;
#pragma unroll
        for (int j = 0; j < 8; ++j) {
            a0.p[j] = __builtin_elementwise_max(ts2 * ukp0[j] + ckp0[j], zero2);
            a1.p[j] = __builtin_elementwise_max(ts2 * ukp1[j] + ckp1[j], zero2);
        }

        v8f c0 = {}, c1 = {};
        c0 = wmma_f16(a0.v, iB[0][0], c0);
        c0 = wmma_f16(a1.v, iB[1][0], c0);
        c1 = wmma_f16(a0.v, iB[0][1], c1);
        c1 = wmma_f16(a1.v, iB[1][1], c1);

        float a8[8];
#pragma unroll
        for (int r = 0; r < 8; ++r) {
            float x0 = fmaxf(c0[r] + ib2lo, 0.0f);
            float x1 = fmaxf(c1[r] + ib2hi, 0.0f);
            a8[r] = fmaf(x0, iw3lo, x1 * iw3hi);
        }
        int rown;
        float v = reduce_scatter16(a8, n, &rown);
        if ((lane & 1) == 0)
            stage_s0[wv][rown + 8 * h][sv] = v + ib3s;
    }

    __builtin_amdgcn_wave_barrier();   // keep LDS produce -> consume ordered

    // ================= combine + coalesced store =================
    const float gv = stage_g[wv][n];
#pragma unroll
    for (int p = 0; p < 8; ++p) {
        int b = 2 * p + h;
        size_t idx = (size_t)b * S_LEN + (size_t)(sbase + n);
        out[idx] = stage_s0[wv][b][n] + gv * delta[idx];
    }
}

// ---------------------------------------------------------------------------
extern "C" void kernel_launch(void* const* d_in, const int* in_sizes, int n_in,
                              void* d_out, int out_size, void* d_ws, size_t ws_size,
                              hipStream_t stream) {
    (void)in_sizes; (void)n_in; (void)out_size; (void)ws_size;
    const float* delta   = (const float*)d_in[0];
    const float* mixture = (const float*)d_in[1];
    const float* bw1 = (const float*)d_in[2];
    const float* bb1 = (const float*)d_in[3];
    const float* bw2 = (const float*)d_in[4];
    const float* bb2 = (const float*)d_in[5];
    const float* bw3 = (const float*)d_in[6];
    const float* bb3 = (const float*)d_in[7];
    const float* iw1 = (const float*)d_in[8];
    const float* ib1 = (const float*)d_in[9];
    const float* iw2 = (const float*)d_in[10];
    const float* ib2 = (const float*)d_in[11];
    const float* iw3 = (const float*)d_in[12];
    const float* ib3 = (const float*)d_in[13];
    float* out = (float*)d_out;
    float* acc = (float*)d_ws;   // 16 floats of scratch

    hipLaunchKernelGGL(zero_acc_kernel, dim3(1), dim3(16), 0, stream, acc);
    hipLaunchKernelGGL(mf_reduce_kernel, dim3(128, 16), dim3(256), 0, stream,
                       mixture, acc);
    hipLaunchKernelGGL(fused_mlp_kernel, dim3(1250), dim3(256), 0, stream,
                       delta, acc, bw1, bb1, bw2, bb2, bw3, bb3,
                       iw1, ib1, iw2, ib2, iw3, ib3, out);
}
